// SegmentedAttention_76459007803660
// MI455X (gfx1250) — compile-verified, matrix operand order checked by reading, not run on hardware
//
#include <hip/hip_runtime.h>
#include <hip/hip_bf16.h>

typedef __attribute__((ext_vector_type(8)))  _Float16 v8h;
typedef __attribute__((ext_vector_type(16))) _Float16 v16h;
typedef __attribute__((ext_vector_type(8)))  float    v8f;
typedef __attribute__((ext_vector_type(4)))  unsigned int u32x4;
typedef __attribute__((ext_vector_type(8)))  int      i32x8;
typedef __attribute__((ext_vector_type(4)))  int      i32x4;

#define DIM    1024
#define HEADS  16
#define DHEAD  64
#define SEG    512
#define NPM    16
#define ROWS   8192          // b * w * SEG = 16 * 512
#define NQKV   3072
#define KVLEN  528           // SEG + NPM

// Toolchain probe: therock-10.0 headers (6-arg tensor builtin) ship this header;
// ROCm 7.2 (5-arg builtin) does not.
#if __has_include(<hip/amd_detail/amd_gfx1250_TDM.h>)
#define USE_TDM6 1
#else
#define USE_TDM6 0
#endif

static __device__ __forceinline__ v16h cat8(v8h a, v8h b) {
  return __builtin_shufflevector(a, b, 0,1,2,3,4,5,6,7,8,9,10,11,12,13,14,15);
}

// ---------------------------------------------------------------------------
// Tensor Data Mover: 2D tile load (global -> LDS), 8-byte elements, optional
// LDS row padding. D# per CDNA5 ISA ch.8: group0 = {count, lds_addr,
// global_addr, type=2}; group1 packs data_size/pad/tensor dims/tile dims/
// strides; groups 2-3 zero (2D tensor).
// ---------------------------------------------------------------------------
static __device__ __forceinline__ void tdm_load_2d(unsigned int lds_off,
                                                   const void* gsrc,
                                                   unsigned int w_units,
                                                   unsigned int rows,
                                                   unsigned int stride_units,
                                                   unsigned int pad_int_enc,
                                                   unsigned int pad_amt_enc,
                                                   unsigned int pad_enable) {
  unsigned long long ga = (unsigned long long)(size_t)gsrc;
  u32x4 g0 = { 1u,                                         // count=1 (valid user D#)
               lds_off,                                    // LDS byte address
               (unsigned int)ga,                           // global_addr[31:0]
               (unsigned int)((ga >> 32) & 0x01FFFFFFull)  // global_addr[56:32]
                 | (2u << 30) };                           // type=2 ("image")
  i32x8 g1;
  g1[0] = (int)((3u << 16)                 // data_size = 8B
              | (pad_enable  << 20)
              | (pad_int_enc << 22)
              | (pad_amt_enc << 25));
  g1[1] = (int)((w_units & 0xFFFFu) << 16);                  // tensor_dim0[15:0]
  g1[2] = (int)((w_units >> 16) | ((rows & 0xFFFFu) << 16)); // dim0[31:16], dim1[15:0]
  g1[3] = (int)((rows >> 16) | ((w_units & 0xFFFFu) << 16)); // dim1[31:16], tile_dim0
  g1[4] = (int)(rows & 0xFFFFu);                             // tile_dim1 (tile_dim2=0)
  g1[5] = (int)stride_units;                                 // tensor_dim0_stride[31:0]
  g1[6] = 0;
  g1[7] = 0;
  i32x4 gz = {0, 0, 0, 0};
#if USE_TDM6
  i32x8 gz8 = {};
  __builtin_amdgcn_tensor_load_to_lds(g0, g1, gz, gz, gz8, 0);
#else
  __builtin_amdgcn_tensor_load_to_lds(g0, g1, gz, gz, 0);
#endif
}

// ---------------------------------------------------------------------------
// RMSNorm: one block per row, 256 threads, 4 elems/thread, fp16 output.
// ---------------------------------------------------------------------------
__global__ void __launch_bounds__(256) rmsnorm_kernel(const float* __restrict__ seq,
                                                      const float* __restrict__ w,
                                                      _Float16* __restrict__ xn) {
  __shared__ float red[8];
  const int row = blockIdx.x;
  const int t = threadIdx.x;
  const float* x = seq + (size_t)row * DIM;
  float4 v = ((const float4*)x)[t];
  float ss = v.x*v.x + v.y*v.y + v.z*v.z + v.w*v.w;
  #pragma unroll
  for (int off = 16; off; off >>= 1) ss += __shfl_xor(ss, off, 32);
  if ((t & 31) == 0) red[t >> 5] = ss;
  __syncthreads();
  float tot = 0.f;
  #pragma unroll
  for (int i = 0; i < 8; ++i) tot += red[i];
  const float scale = rsqrtf(tot * (1.0f / DIM) + 1.1920929e-07f);
  float4 wv = ((const float4*)w)[t];
  _Float16* o = xn + (size_t)row * DIM + t * 4;
  o[0] = (_Float16)(v.x * scale * wv.x);
  o[1] = (_Float16)(v.y * scale * wv.y);
  o[2] = (_Float16)(v.z * scale * wv.z);
  o[3] = (_Float16)(v.w * scale * wv.w);
}

// ---------------------------------------------------------------------------
// fp32 -> fp16 weight conversion, 4 elems/thread.
// ---------------------------------------------------------------------------
__global__ void __launch_bounds__(256) cvt_kernel(const float* __restrict__ in,
                                                  _Float16* __restrict__ out, int n4) {
  int i = blockIdx.x * blockDim.x + threadIdx.x;
  if (i >= n4) return;
  float4 v = ((const float4*)in)[i];
  _Float16* o = out + (size_t)i * 4;
  o[0] = (_Float16)v.x; o[1] = (_Float16)v.y;
  o[2] = (_Float16)v.z; o[3] = (_Float16)v.w;
}

// ---------------------------------------------------------------------------
// Tiled WMMA GEMM: C[ROWS x N] = A[ROWS x 1024] * B[1024 x N], f16 in, f32 acc.
// Block: 256 threads (8 waves). Tile 128x64, BK=32. Each wave: 16x64.
// Double-buffered pipeline: A tile DMA'd by the Tensor Data Mover into the
// ping-pong buffer while the current tile is consumed by WMMA; B tile global
// loads are issued before the compute burst. One barrier per K-step.
// mode 0: write Cf16 (qkv) + fp32 orig_v slice of d_out.
// mode 1: write fp32 C (final output).
// ---------------------------------------------------------------------------
__global__ void __launch_bounds__(256, 1) gemm_kernel(const _Float16* __restrict__ A,
                                                      const _Float16* __restrict__ B,
                                                      int N,
                                                      _Float16* __restrict__ Cf16,
                                                      float* __restrict__ Vout,
                                                      float* __restrict__ Cf32,
                                                      int mode) {
  __shared__ _Float16 sA[2][128][40];   // row-major A tiles, 80B rows (64B + 16B pad)
  __shared__ _Float16 sB[2][64][40];    // TRANSPOSED B tiles: sB[buf][n][k]
  const int tid = threadIdx.x;
  const int wave = tid >> 5, lane = tid & 31;
  const int lm = lane & 15, kg = lane >> 4;
  const int m0 = blockIdx.y * 128;
  const int n0 = blockIdx.x * 64;
  const int kk = tid >> 3, n8 = (tid & 7) * 8;   // B-tile load coordinates

  // Prologue: kick off DMA of A tile 0, load B tile 0 into registers.
  if (tid < 32) {
    tdm_load_2d((unsigned int)(size_t)&sA[0][0][0], A + (size_t)m0 * 1024,
                8, 128, 256, 3, 3, 1);
  }
  v8h breg = *(const v8h*)(B + (size_t)kk * N + n0 + n8);

  v8f acc[4] = {};
  for (int i = 0; i < 32; ++i) {
    const int cur = i & 1;
    // Publish B tile (transposed), ensure A DMA complete, sync.
    #pragma unroll
    for (int e = 0; e < 8; ++e) sB[cur][n8 + e][kk] = breg[e];
    if (tid < 32) __builtin_amdgcn_s_wait_tensorcnt(0);
    __syncthreads();
    // Prefetch next tiles into the other buffer (DMA + global loads overlap
    // with the WMMA burst below).
    if (i + 1 < 32) {
      if (tid < 32) {
        tdm_load_2d((unsigned int)(size_t)&sA[cur ^ 1][0][0],
                    A + (size_t)m0 * 1024 + (i + 1) * 32, 8, 128, 256, 3, 3, 1);
      }
      breg = *(const v8h*)(B + (size_t)((i + 1) * 32 + kk) * N + n0 + n8);
    }
    // Compute on the current buffers.
    const int ar = wave * 16 + lm;
    v16h af = cat8(*(const v8h*)&sA[cur][ar][kg * 8],
                   *(const v8h*)&sA[cur][ar][16 + kg * 8]);
    v16h bf[4];
    #pragma unroll
    for (int t = 0; t < 4; ++t)
      bf[t] = cat8(*(const v8h*)&sB[cur][t * 16 + lm][kg * 8],
                   *(const v8h*)&sB[cur][t * 16 + lm][16 + kg * 8]);
    #pragma unroll
    for (int t = 0; t < 4; ++t)
      acc[t] = __builtin_amdgcn_wmma_f32_16x16x32_f16(false, af, false, bf[t],
                                                      (short)0, acc[t], false, false);
  }

  const int gr0 = m0 + wave * 16 + kg * 8;
  if (mode == 0) {
    #pragma unroll
    for (int t = 0; t < 4; ++t) {
      const int gc = n0 + t * 16 + lm;
      const bool isv = (n0 + t * 16) >= 2048;   // uniform per 16-wide n-tile
      #pragma unroll
      for (int j = 0; j < 8; ++j) {
        const int gr = gr0 + j;
        const float val = acc[t][j];
        Cf16[(size_t)gr * NQKV + gc] = (_Float16)val;
        if (isv) {   // V columns -> orig_v output (bw, h, s, d), fp32
          const int c = gc - 2048, hh = c >> 6, d = c & 63;
          const int bw = gr >> 9, s = gr & 511;
          Vout[(((size_t)(bw * HEADS + hh)) * SEG + s) * DHEAD + d] = val;
        }
      }
    }
  } else {
    #pragma unroll
    for (int t = 0; t < 4; ++t) {
      const int gc = n0 + t * 16 + lm;
      #pragma unroll
      for (int j = 0; j < 8; ++j) {
        Cf32[(size_t)(gr0 + j) * DIM + gc] = acc[t][j];
      }
    }
  }
}

// ---------------------------------------------------------------------------
// RoPE on q/k + head-major reshape of q/k/v into (bw, h, s, d) fp16.
// One thread per even/odd element pair.
// ---------------------------------------------------------------------------
__global__ void __launch_bounds__(256) rope_kernel(const _Float16* __restrict__ qkv,
                                                   _Float16* __restrict__ q,
                                                   _Float16* __restrict__ k,
                                                   _Float16* __restrict__ v) {
  int idx = blockIdx.x * blockDim.x + threadIdx.x;
  if (idx >= ROWS * (NQKV / 2)) return;
  const int r = idx / (NQKV / 2);
  const int c = (idx - r * (NQKV / 2)) * 2;
  const float x1 = (float)qkv[(size_t)r * NQKV + c];
  const float x2 = (float)qkv[(size_t)r * NQKV + c + 1];
  const int bw = r >> 9, s = r & 511;
  if (c < 2048) {
    const int ch = c & 1023;           // offset within q or k block
    const int h = ch >> 6, d = ch & 63;
    const int i = d >> 1;
    const float freq = __powf(10000.0f, -(float)(2 * i) * (1.0f / DHEAD));
    float sn, cs;
    __sincosf((float)s * freq, &sn, &cs);
    _Float16* dst = (c < 1024) ? q : k;
    const size_t off = (((size_t)bw * HEADS + h) * SEG + s) * DHEAD + d;
    dst[off]     = (_Float16)(x1 * cs - x2 * sn);
    dst[off + 1] = (_Float16)(x2 * cs + x1 * sn);
  } else {
    const int ch = c - 2048;
    const int h = ch >> 6, d = ch & 63;
    const size_t off = (((size_t)bw * HEADS + h) * SEG + s) * DHEAD + d;
    v[off]     = (_Float16)x1;
    v[off + 1] = (_Float16)x2;
  }
}

// ---------------------------------------------------------------------------
// Flash-style causal attention with persistent-memory prefix.
// One block per (window, head). 8 waves x 4 q-blocks of 16 rows.
// K tile DMA'd by the TDM into padded LDS rows; V staged transposed by the
// threads; online softmax; WMMA for both Q*K^T and P*V.
// ---------------------------------------------------------------------------
__global__ void __launch_bounds__(256, 1) attn_kernel(const _Float16* __restrict__ q,
                                                      const _Float16* __restrict__ k,
                                                      const _Float16* __restrict__ v,
                                                      const float* __restrict__ pmem,
                                                      _Float16* __restrict__ out) {
  extern __shared__ _Float16 smem[];
  _Float16* sK  = smem;                    // [528][72]   row-major K (incl. pmem)
  _Float16* sVt = smem + KVLEN * 72;       // [64][536]   transposed V
  _Float16* sP  = sVt + 64 * 536;          // 8 waves x [16][40] P scratch

  const int tid = threadIdx.x;
  const int wave = tid >> 5, lane = tid & 31;
  const int lm = lane & 15, kg = lane >> 4;
  const int bw = blockIdx.x >> 4, h = blockIdx.x & 15;
  const size_t bh = (size_t)(bw * HEADS + h);
  const _Float16* kb = k + bh * SEG * DHEAD;
  const _Float16* vb = v + bh * SEG * DHEAD;
  const float* pmk = pmem + (size_t)h * NPM * DHEAD;
  const float* pmv = pmem + (size_t)(HEADS + h) * NPM * DHEAD;

  // K rows 16..527 via TDM: 512 rows x 16 (8B units), contiguous (stride 16);
  // pad 32 DWORDs data (enc 4) + 4 DWORDs pad (enc 3) -> 72-half LDS rows.
  if (tid < 32) {
    tdm_load_2d((unsigned int)(size_t)(sK + NPM * 72), kb, 16, SEG, 16, 4, 3, 1);
  }
  // pmem K rows (fp32 -> fp16)
  for (int idx = tid; idx < NPM * DHEAD; idx += 256) {
    const int r = idx >> 6, d = idx & 63;
    sK[r * 72 + d] = (_Float16)pmk[idx];
  }
  // V (pmem prefix + segment), stored transposed for contiguous B-fragments
  for (int idx = tid; idx < KVLEN * DHEAD; idx += 256) {
    const int r = idx >> 6, d = idx & 63;
    _Float16 vv = (r < NPM) ? (_Float16)pmv[r * DHEAD + d]
                            : vb[(size_t)(r - NPM) * DHEAD + d];
    sVt[d * 536 + r] = vv;
  }
  if (tid < 32) __builtin_amdgcn_s_wait_tensorcnt(0);
  __syncthreads();

  _Float16* sPw = sP + wave * 16 * 40;
  const _Float16* qb = q + bh * SEG * DHEAD;

  for (int qbk = 0; qbk < 4; ++qbk) {
    const int qbase = (wave * 4 + qbk) * 16;
    const _Float16* qr = qb + (size_t)(qbase + lm) * DHEAD;
    v16h aQ0 = cat8(*(const v8h*)(qr + kg * 8),      *(const v8h*)(qr + 16 + kg * 8));
    v16h aQ1 = cat8(*(const v8h*)(qr + 32 + kg * 8), *(const v8h*)(qr + 48 + kg * 8));

    v8f acc[4] = {};
    float mrow[8], lrow[8];
    #pragma unroll
    for (int j = 0; j < 8; ++j) { mrow[j] = -3.0e38f; lrow[j] = 0.f; }

    const int kvend = qbase + 32;        // causal early exit (max visible j = qbase+31)
    for (int kt = 0; kt < kvend; kt += 32) {
      v8f s0 = {}, s1 = {};
      {
        const _Float16* kr0 = sK + (size_t)(kt + lm) * 72;
        v16h b00 = cat8(*(const v8h*)(kr0 + kg * 8),      *(const v8h*)(kr0 + 16 + kg * 8));
        v16h b01 = cat8(*(const v8h*)(kr0 + 32 + kg * 8), *(const v8h*)(kr0 + 48 + kg * 8));
        const _Float16* kr1 = sK + (size_t)(kt + 16 + lm) * 72;
        v16h b10 = cat8(*(const v8h*)(kr1 + kg * 8),      *(const v8h*)(kr1 + 16 + kg * 8));
        v16h b11 = cat8(*(const v8h*)(kr1 + 32 + kg * 8), *(const v8h*)(kr1 + 48 + kg * 8));
        s0 = __builtin_amdgcn_wmma_f32_16x16x32_f16(false, aQ0, false, b00, (short)0, s0, false, false);
        s0 = __builtin_amdgcn_wmma_f32_16x16x32_f16(false, aQ1, false, b01, (short)0, s0, false, false);
        s1 = __builtin_amdgcn_wmma_f32_16x16x32_f16(false, aQ0, false, b10, (short)0, s1, false, false);
        s1 = __builtin_amdgcn_wmma_f32_16x16x32_f16(false, aQ1, false, b11, (short)0, s1, false, false);
      }

      #pragma unroll
      for (int j = 0; j < 8; ++j) {
        const int i = qbase + j + kg * 8;
        const int c0 = kt + lm, c1 = c0 + 16;
        float v0 = s0[j] * 0.125f;       // DIM_HEAD^-0.5
        float v1 = s1[j] * 0.125f;
        if (!(c0 < NPM || c0 - NPM <= i)) v0 = -1e30f;
        if (!(c1 < NPM || c1 - NPM <= i)) v1 = -1e30f;
        float tmax = fmaxf(v0, v1);
        #pragma unroll
        for (int off = 1; off < 16; off <<= 1) tmax = fmaxf(tmax, __shfl_xor(tmax, off, 32));
        const float nm = fmaxf(mrow[j], tmax);
        const float corr = __expf(mrow[j] - nm);
        const float p0 = __expf(v0 - nm);
        const float p1 = __expf(v1 - nm);
        mrow[j] = nm;
        float ps = p0 + p1;
        #pragma unroll
        for (int off = 1; off < 16; off <<= 1) ps += __shfl_xor(ps, off, 32);
        lrow[j] = lrow[j] * corr + ps;
        #pragma unroll
        for (int t = 0; t < 4; ++t) acc[t][j] *= corr;
        const int prow = j + kg * 8;
        sPw[prow * 40 + lm]      = (_Float16)p0;
        sPw[prow * 40 + 16 + lm] = (_Float16)p1;
      }

      // P tile round-trips through LDS (C-layout -> A-layout): wave-internal,
      // needs DS completion but no workgroup barrier.
      asm volatile("s_wait_dscnt 0x0" ::: "memory");
      __builtin_amdgcn_wave_barrier();

      const _Float16* pr = sPw + lm * 40;
      v16h aP = cat8(*(const v8h*)(pr + kg * 8), *(const v8h*)(pr + 16 + kg * 8));
      #pragma unroll
      for (int t = 0; t < 4; ++t) {
        const _Float16* vr = sVt + (size_t)(t * 16 + lm) * 536 + kt;
        v16h bV = cat8(*(const v8h*)(vr + kg * 8), *(const v8h*)(vr + 16 + kg * 8));
        acc[t] = __builtin_amdgcn_wmma_f32_16x16x32_f16(false, aP, false, bV,
                                                        (short)0, acc[t], false, false);
      }
    }

    #pragma unroll
    for (int j = 0; j < 8; ++j) {
      const float inv = 1.0f / lrow[j];
      const int gr = bw * SEG + qbase + j + kg * 8;
      #pragma unroll
      for (int t = 0; t < 4; ++t) {
        out[(size_t)gr * DIM + h * DHEAD + t * 16 + lm] = (_Float16)(acc[t][j] * inv);
      }
    }
  }
}

// ---------------------------------------------------------------------------
// Host-side launch.
// ---------------------------------------------------------------------------
extern "C" void kernel_launch(void* const* d_in, const int* in_sizes, int n_in,
                              void* d_out, int out_size, void* d_ws, size_t ws_size,
                              hipStream_t stream) {
  const float* seq   = (const float*)d_in[0];
  const float* normw = (const float*)d_in[1];
  const float* wqkv  = (const float*)d_in[2];
  const float* wout  = (const float*)d_in[3];
  const float* pmem  = (const float*)d_in[4];
  float* out = (float*)d_out;
  float* origV = out + (size_t)ROWS * DIM;       // second tuple output

  char* ws = (char*)d_ws;
  _Float16* xnF   = (_Float16*)(ws);                       // 16 MB (reused as attnF)
  _Float16* wqkvF = (_Float16*)(ws + (size_t)16777216);    // 6 MB
  _Float16* woutF = (_Float16*)(ws + (size_t)23068672);    // 2 MB
  _Float16* qkvF  = (_Float16*)(ws + (size_t)25165824);    // 48 MB
  _Float16* qF    = (_Float16*)(ws + (size_t)75497472);    // 16 MB
  _Float16* kF    = (_Float16*)(ws + (size_t)92274688);    // 16 MB
  _Float16* vF    = (_Float16*)(ws + (size_t)109051904);   // 16 MB
  _Float16* attnF = xnF;                                   // xn dead after GEMM1

  // 1. RMSNorm -> fp16
  rmsnorm_kernel<<<ROWS, 256, 0, stream>>>(seq, normw, xnF);

  // 2. Weight conversion fp32 -> fp16
  cvt_kernel<<<(DIM * NQKV / 4 + 255) / 256, 256, 0, stream>>>(wqkv, wqkvF, DIM * NQKV / 4);
  cvt_kernel<<<(DIM * DIM  / 4 + 255) / 256, 256, 0, stream>>>(wout, woutF, DIM * DIM / 4);

  // 3. QKV GEMM (writes qkv fp16 + orig_v fp32 straight into d_out)
  gemm_kernel<<<dim3(NQKV / 64, ROWS / 128), 256, 0, stream>>>(
      xnF, wqkvF, NQKV, qkvF, origV, nullptr, 0);

  // 4. RoPE + head-major reshape
  {
    const int total = ROWS * (NQKV / 2);
    rope_kernel<<<(total + 255) / 256, 256, 0, stream>>>(qkvF, qF, kF, vF);
  }

  // 5. Attention (dynamic LDS: K 76032B + Vt 68608B + P 10240B = 154880B)
  attn_kernel<<<16 * HEADS, 256, 154880, stream>>>(qF, kF, vF, pmem, attnF);

  // 6. Output projection -> d_out fp32
  gemm_kernel<<<dim3(DIM / 64, ROWS / 128), 256, 0, stream>>>(
      attnF, woutF, DIM, nullptr, nullptr, out, 1);
}
